// RNNBase_40905268527143
// MI455X (gfx1250) — matrix operand order repaired and checked by reading, not compiled, not run on hardware
//
#include <hip/hip_runtime.h>

// ---------------------------------------------------------------------------
// 2-layer masked LSTM for MI455X (gfx1250, wave32, WMMA bf16 16x16x32)
// ---------------------------------------------------------------------------

#define Bsz 64
#define Tsz 512
#define INsz 256
#define Hsz 512

typedef __attribute__((ext_vector_type(16))) __bf16 v16bf;
typedef __attribute__((ext_vector_type(8)))  float  v8f;

__device__ __forceinline__ unsigned short f2bf(float f) {
  unsigned int u = __float_as_uint(f);
  return (unsigned short)((u + 0x7FFFu + ((u >> 16) & 1u)) >> 16);  // RNE
}

__device__ __forceinline__ float sigf(float x) {
  return 1.0f / (1.0f + __expf(-x));
}

// ---------------------------------------------------------------------------
// Pack x[B,T,IN] (fp32) into WMMA A-fragment bf16 layout, per t:
//   xA[t][mt(4)][kt(8)][lane(32)][half(16)]
// A 16x32 bf16 layout (ISA 7.12.2): lane<16: halves 0..7 -> K=j, 8..15 -> K=j+8;
//                                    lane>=16: halves 0..7 -> K=j+8, 8..15 -> K=j+16
// ---------------------------------------------------------------------------
__global__ void pack_x_kernel(const float* __restrict__ x,
                              unsigned short* __restrict__ xA) {
  int i = blockIdx.x * blockDim.x + threadIdx.x;   // T*4*8*32*16 = 8388608
  int j    = i & 15;
  int lane = (i >> 4) & 31;
  int kt   = (i >> 9) & 7;
  int mt   = (i >> 12) & 3;
  int t    = i >> 14;
  int M = lane & 15;
  int K = (j < 8 ? j : j + 8) + ((lane >= 16) ? 8 : 0);
  int b = mt * 16 + M;
  int k = kt * 32 + K;
  xA[i] = f2bf(x[((size_t)b * Tsz + t) * INsz + k]);
}

// ---------------------------------------------------------------------------
// Pack W[4H, Kdim] (fp32) into WMMA B-fragment bf16 layout:
//   WB[nt(128)][kt(ktW)][lane(32)][half(16)]
// B 32x16 bf16 layout: N = lane&15; same K order as A: lane<16: j<8 -> K=j, else K=j+8; lane>=16: +8
// ---------------------------------------------------------------------------
__global__ void pack_w_kernel(const float* __restrict__ W,
                              unsigned short* __restrict__ WB,
                              int ktW, int Kdim) {
  int i = blockIdx.x * blockDim.x + threadIdx.x;   // 128*ktW*512
  int j    = i & 15;
  int lane = (i >> 4) & 31;
  int kt   = (i >> 9) % ktW;
  int nt   = i / (ktW << 9);
  int N = nt * 16 + (lane & 15);
  int K = kt * 32 + (j < 8 ? j : j + 8) + ((lane >= 16) ? 8 : 0);
  WB[i] = f2bf(W[(size_t)N * Kdim + K]);
}

// ---------------------------------------------------------------------------
// One fused LSTM time-step: gates = inA*WxB + hA*WhB + b ; cell ; mask ;
// write fp32 state + bf16 A-fragment-packed h for the next step.
// grid = 16 blocks x 256 threads = 128 waves; wave w: mt = w>>5, ng = w&31.
// Wave owns 16x16 output tile for each of the 4 gates (cols ng*16 + 512*g).
// ---------------------------------------------------------------------------
__global__ __launch_bounds__(256) void lstm_step_kernel(
    const unsigned short* __restrict__ inA, int ktx,         // input A frags
    const unsigned short* __restrict__ WxB,                  // input weights (B frags)
    const unsigned short* __restrict__ hA,                   // recurrent A frags (16 K-tiles)
    const unsigned short* __restrict__ WhB,                  // recurrent weights
    const float* __restrict__ bih, const float* __restrict__ bhh,
    float* __restrict__ hbuf, float* __restrict__ cbuf,      // [B,H] fp32 state (in place)
    const float* __restrict__ resetH, const float* __restrict__ resetC,
    const int* __restrict__ length, int t,
    unsigned short* __restrict__ hA_out,                     // packed bf16 h_next
    float* __restrict__ hs_out, float* __restrict__ last_out,
    int layer) {
  int tid  = threadIdx.x;
  int lane = tid & 31;
  int w    = blockIdx.x * 8 + (tid >> 5);
  int mt   = w >> 5;   // batch tile 0..3
  int ng   = w & 31;   // hidden group 0..31

  v8f acc[4];
  v8f z = {0.f, 0.f, 0.f, 0.f, 0.f, 0.f, 0.f, 0.f};
#pragma unroll
  for (int g = 0; g < 4; ++g) acc[g] = z;

  // ---- input GEMM: K = 32*ktx ----
  for (int kt = 0; kt < ktx; ++kt) {
    v16bf a = *(const v16bf*)(inA + (((size_t)mt * ktx + kt) * 32 + lane) * 16);
#pragma unroll
    for (int g = 0; g < 4; ++g) {
      int nt = ng + 32 * g;
      v16bf b = *(const v16bf*)(WxB + (((size_t)nt * ktx + kt) * 32 + lane) * 16);
      acc[g] = __builtin_amdgcn_wmma_f32_16x16x32_bf16(false, a, false, b,
                                                       (short)0, acc[g], false, false);
    }
  }
  // ---- recurrent GEMM: K = 512 (16 tiles) ----
  for (int kt = 0; kt < 16; ++kt) {
    v16bf a = *(const v16bf*)(hA + (((size_t)mt * 16 + kt) * 32 + lane) * 16);
#pragma unroll
    for (int g = 0; g < 4; ++g) {
      int nt = ng + 32 * g;
      v16bf b = *(const v16bf*)(WhB + (((size_t)nt * 16 + kt) * 32 + lane) * 16);
      acc[g] = __builtin_amdgcn_wmma_f32_16x16x32_bf16(false, a, false, b,
                                                       (short)0, acc[g], false, false);
    }
  }

  // ---- LSTM cell, in registers (C/D layout: M = r + 8*lane[4], N = lane&15) ----
  int col    = ng * 16 + (lane & 15);
  int rowadd = (lane >= 16) ? 8 : 0;
#pragma unroll
  for (int r = 0; r < 8; ++r) {
    int brow = mt * 16 + r + rowadd;
    float gi = acc[0][r] + bih[col]        + bhh[col];
    float gf = acc[1][r] + bih[col + 512]  + bhh[col + 512];
    float gg = acc[2][r] + bih[col + 1024] + bhh[col + 1024];
    float go = acc[3][r] + bih[col + 1536] + bhh[col + 1536];
    size_t si = (size_t)brow * Hsz + col;
    float cp = cbuf[si];
    float cn = sigf(gf) * cp + sigf(gi) * tanhf(gg);
    float hn = sigf(go) * tanhf(cn);
    bool valid = (t < length[brow]);
    if (!valid) {
      if (layer == 0) { hn = hbuf[si]; cn = cp; }          // keep carry
      else            { hn = resetH[si]; cn = resetC[si]; } // reset to layer-0 finals
    }
    cbuf[si] = cn;
    hbuf[si] = hn;
    // pack h_next into A-fragment layout for the next step's WMMA
    int kt = col >> 5, K = col & 31, M = brow & 15, lp, jp;
    if      (K < 8)  { lp = M;      jp = K;      }
    else if (K < 16) { lp = M + 16; jp = K - 8;  }
    else if (K < 24) { lp = M;      jp = K - 8;  }
    else             { lp = M + 16; jp = K - 16; }
    hA_out[(((size_t)mt * 16 + kt) * 32 + lp) * 16 + jp] = f2bf(hn);
    if (hs_out) {
      hs_out[((size_t)brow * Tsz + t) * Hsz + col] = hn;
      if (brow == Bsz - 1) last_out[(size_t)t * Hsz + col] = hn;
    }
  }
}

// ---------------------------------------------------------------------------
extern "C" void kernel_launch(void* const* d_in, const int* in_sizes, int n_in,
                              void* d_out, int out_size, void* d_ws, size_t ws_size,
                              hipStream_t stream) {
  (void)in_sizes; (void)n_in; (void)out_size; (void)ws_size;
  const float* x     = (const float*)d_in[0];
  const int*   length= (const int*)  d_in[1];
  const float* Wih0  = (const float*)d_in[2];
  const float* Whh0  = (const float*)d_in[3];
  const float* bih0  = (const float*)d_in[4];
  const float* bhh0  = (const float*)d_in[5];
  const float* Wih1  = (const float*)d_in[6];
  const float* Whh1  = (const float*)d_in[7];
  const float* bih1  = (const float*)d_in[8];
  const float* bhh1  = (const float*)d_in[9];
  float* out = (float*)d_out;

  // ---- workspace carve-out (~56 MB) ----
  char* p = (char*)d_ws;
  auto take = [&](size_t bytes) {
    char* r = p; p += (bytes + 255) & ~(size_t)255; return r;
  };
  const size_t AS  = (size_t)4 * 16 * 32 * 16;                // 32768 bf16 = one packed [B,H] A
  const size_t XAS = (size_t)4 * 8  * 32 * 16;                // one packed [B,IN] A
  unsigned short* xA    = (unsigned short*)take((size_t)Tsz * XAS * 2);   // 16 MB
  unsigned short* hs0A  = (unsigned short*)take((size_t)Tsz * AS  * 2);   // 32 MB
  unsigned short* Wih0B = (unsigned short*)take((size_t)128 * 8  * 512 * 2);
  unsigned short* Whh0B = (unsigned short*)take((size_t)128 * 16 * 512 * 2);
  unsigned short* Wih1B = (unsigned short*)take((size_t)128 * 16 * 512 * 2);
  unsigned short* Whh1B = (unsigned short*)take((size_t)128 * 16 * 512 * 2);
  float* hbuf0 = (float*)take((size_t)Bsz * Hsz * 4);
  float* cbuf0 = (float*)take((size_t)Bsz * Hsz * 4);
  float* hbuf1 = (float*)take((size_t)Bsz * Hsz * 4);
  float* cbuf1 = (float*)take((size_t)Bsz * Hsz * 4);
  unsigned short* hA1 = (unsigned short*)take(2 * AS * 2);    // double buffer
  unsigned short* zA  = (unsigned short*)take(AS * 2);        // zero A for layer0 t=0

  // zero-init state (contiguous thanks to 256B-aligned, 256B-multiple sizes)
  hipMemsetAsync(hbuf0, 0, (size_t)4 * Bsz * Hsz * 4, stream);
  hipMemsetAsync(hA1,   0, (2 * AS + AS) * 2, stream);

  // ---- pack inputs & weights into WMMA fragment layouts ----
  pack_x_kernel<<<(Tsz * XAS) / 256, 256, 0, stream>>>(x, xA);
  pack_w_kernel<<<((size_t)128 * 8  * 512) / 256, 256, 0, stream>>>(Wih0, Wih0B, 8,  INsz);
  pack_w_kernel<<<((size_t)128 * 16 * 512) / 256, 256, 0, stream>>>(Whh0, Whh0B, 16, Hsz);
  pack_w_kernel<<<((size_t)128 * 16 * 512) / 256, 256, 0, stream>>>(Wih1, Wih1B, 16, Hsz);
  pack_w_kernel<<<((size_t)128 * 16 * 512) / 256, 256, 0, stream>>>(Whh1, Whh1B, 16, Hsz);

  // ---- layer 0: masked steps keep carry; hs0 written packed (feeds layer 1) ----
  for (int t = 0; t < Tsz; ++t) {
    const unsigned short* hprev = (t == 0) ? zA : (hs0A + (size_t)(t - 1) * AS);
    lstm_step_kernel<<<16, 256, 0, stream>>>(
        xA + (size_t)t * XAS, 8, Wih0B, hprev, Whh0B,
        bih0, bhh0, hbuf0, cbuf0, nullptr, nullptr, length, t,
        hs0A + (size_t)t * AS, nullptr, nullptr, 0);
  }

  // ---- layer 1: masked steps reset to layer-0 FINAL state (hbuf0/cbuf0) ----
  for (int t = 0; t < Tsz; ++t) {
    lstm_step_kernel<<<16, 256, 0, stream>>>(
        hs0A + (size_t)t * AS, 16, Wih1B,
        hA1 + (size_t)(t & 1) * AS, Whh1B,
        bih1, bhh1, hbuf1, cbuf1, hbuf0, cbuf0, length, t,
        hA1 + (size_t)((t + 1) & 1) * AS,
        out, out + (size_t)Bsz * Tsz * Hsz, 1);
  }
}
